// PointGNN_25537875542451
// MI455X (gfx1250) — matrix operand order repaired and checked by reading, not compiled, not run on hardware
//
#include <hip/hip_runtime.h>

#define LL 3
#define SD 8
#define BB 2
#define NN 512
#define RR 0.05f
#define EPS 1e-5f

typedef _Float16 f16;
typedef __attribute__((ext_vector_type(16))) _Float16 v16h;
typedef __attribute__((ext_vector_type(8)))  _Float16 v8h;
typedef __attribute__((ext_vector_type(8)))  float    v8f;

#define LDS_FENCE() asm volatile("s_wait_dscnt 0x0" ::: "memory")

__device__ __forceinline__ v8f wmma_f16(v16h a, v16h b, v8f c) {
  // v_wmma_f32_16x16x32_f16  D = A*B + C
  return __builtin_amdgcn_wmma_f32_16x16x32_f16(false, a, false, b, (short)0, c, false, false);
}

__device__ __forceinline__ v8f vzero8() {
  v8f r;
#pragma unroll
  for (int q = 0; q < 8; ++q) r[q] = 0.f;
  return r;
}

// A operand load (16x32 f16 tile in LDS, row-major, row stride = KIN halves).
// Caller passes rowbase = buf + row*KIN + ks*32 + hfl*8.
// lane<16 : K 0..7 at +0, K16..23 at +16 ; lane>=16 : K 8..15 / 24..31.
__device__ __forceinline__ v16h ldA(const f16* rowbase) {
  v8h lo = *(const v8h*)(rowbase);
  v8h hi = *(const v8h*)(rowbase + 16);
  v16h r;
#pragma unroll
  for (int q = 0; q < 8; ++q) { r[q] = lo[q]; r[q + 8] = hi[q]; }
  return r;
}

// B operand load: weights [NOUT][KIN] row-major f16 in LDS.
// lane = output column; caller passes +hfl*8: lane<16 gets K0..7/K16..23, lane>=16 gets K8..15/K24..31.
__device__ __forceinline__ v16h ldB(const f16* p) {
  v8h lo = *(const v8h*)(p);
  v8h hi = *(const v8h*)(p + 16);
  v16h r;
#pragma unroll
  for (int q = 0; q < 8; ++q) { r[q] = lo[q]; r[q + 8] = hi[q]; }
  return r;
}

// bias + LayerNorm + ReLU on D tiles (M=16 edges, N=NT*16 feats), store f16 [16][NOUT] to LDS.
template <int NT>
__device__ __forceinline__ void ln_relu_store(v8f* acc, const float* bias, const float* gamma,
                                              const float* beta, f16* Out, int lane) {
  const int NOUT = NT * 16;
  const int col = lane & 15;
  const int hfl = lane >> 4;
  float s1[8], s2[8];
#pragma unroll
  for (int r = 0; r < 8; ++r) { s1[r] = 0.f; s2[r] = 0.f; }
#pragma unroll
  for (int nt = 0; nt < NT; ++nt) {
    float bf = bias[nt * 16 + col];
#pragma unroll
    for (int r = 0; r < 8; ++r) {
      float v = acc[nt][r] + bf;
      acc[nt][r] = v;
      s1[r] += v;
      s2[r] += v * v;
    }
  }
  // reduce the 16 feature-columns held across the lane half (xor 1,2,4,8 stays in-half)
#pragma unroll
  for (int r = 0; r < 8; ++r) {
#pragma unroll
    for (int m = 1; m < 16; m <<= 1) {
      s1[r] += __shfl_xor(s1[r], m, 32);
      s2[r] += __shfl_xor(s2[r], m, 32);
    }
  }
  float mu[8], rs[8];
  const float inv = 1.f / (float)NOUT;
#pragma unroll
  for (int r = 0; r < 8; ++r) {
    mu[r] = s1[r] * inv;
    float var = s2[r] * inv - mu[r] * mu[r];
    rs[r] = rsqrtf(var + EPS);
  }
#pragma unroll
  for (int nt = 0; nt < NT; ++nt) {
    float g = gamma[nt * 16 + col], be = beta[nt * 16 + col];
#pragma unroll
    for (int r = 0; r < 8; ++r) {
      float y = (acc[nt][r] - mu[r]) * rs[r] * g + be;
      y = fmaxf(y, 0.f);
      Out[(hfl * 8 + r) * NOUT + nt * 16 + col] = (f16)y;  // edge = r + 8*half
    }
  }
}

// ---------------- weight conversion: f32 -> f16 (f_w1 zero-padded 11 -> 32) ---------------
__global__ void kcvt_PointGNN(const float* __restrict__ fw1, const float* __restrict__ fw2,
                              const float* __restrict__ fw3, f16* __restrict__ w1o,
                              f16* __restrict__ w2o, f16* __restrict__ w3o) {
  int idx = blockIdx.x * blockDim.x + threadIdx.x;
  if (idx < LL * 64 * 32) {
    int t = idx / (64 * 32), rem = idx % (64 * 32), o = rem / 32, k = rem % 32;
    w1o[idx] = (f16)((k < 11) ? fw1[(t * 64 + o) * 11 + k] : 0.f);
  }
  idx -= LL * 64 * 32;
  if (idx >= 0 && idx < LL * 128 * 64) w2o[idx] = (f16)fw2[idx];
  idx -= LL * 128 * 64;
  if (idx >= 0 && idx < LL * 128 * 128) w3o[idx] = (f16)fw3[idx];
}

// ---------------- per-node: delta = mlp_h(state); emit z_j = [x,s,0..] and u_i = [delta-x,0..]
__global__ __launch_bounds__(32) void knode_PointGNN(
    const float* __restrict__ cur, const float* __restrict__ orig,
    const float* __restrict__ hw1, const float* __restrict__ hb1, const float* __restrict__ hg1,
    const float* __restrict__ hbe1, const float* __restrict__ hw2, const float* __restrict__ hb2,
    const float* __restrict__ hg2, const float* __restrict__ hbe2, const float* __restrict__ hw3,
    const float* __restrict__ hb3, f16* __restrict__ zf, f16* __restrict__ uf, int t) {
  __shared__ float h1[64];
  __shared__ float h2[128];
  const int node = blockIdx.x;  // 0..B*N-1
  const int lane = threadIdx.x;
  const float* si = cur + node * SD;
  const float* xi = orig + node * SD;

  // layer1: 8 -> 64 (2 outs/lane) + LN + relu
  const float* W1 = hw1 + t * 64 * 8;
  float a0[2];
#pragma unroll
  for (int p = 0; p < 2; ++p) {
    int o = lane + p * 32;
    float acc = hb1[t * 64 + o];
#pragma unroll
    for (int k = 0; k < 8; ++k) acc += W1[o * 8 + k] * si[k];
    a0[p] = acc;
  }
  float s1 = a0[0] + a0[1], s2 = a0[0] * a0[0] + a0[1] * a0[1];
  for (int m = 1; m < 32; m <<= 1) { s1 += __shfl_xor(s1, m, 32); s2 += __shfl_xor(s2, m, 32); }
  float mu = s1 / 64.f, var = s2 / 64.f - mu * mu, rs = rsqrtf(var + EPS);
#pragma unroll
  for (int p = 0; p < 2; ++p) {
    int o = lane + p * 32;
    h1[o] = fmaxf((a0[p] - mu) * rs * hg1[t * 64 + o] + hbe1[t * 64 + o], 0.f);
  }
  __syncthreads();

  // layer2: 64 -> 128 (4 outs/lane) + LN + relu
  const float* W2 = hw2 + t * 128 * 64;
  float b4[4];
  float t1 = 0.f, t2 = 0.f;
#pragma unroll
  for (int p = 0; p < 4; ++p) {
    int o = lane + p * 32;
    float acc = hb2[t * 128 + o];
    for (int k = 0; k < 64; ++k) acc += W2[o * 64 + k] * h1[k];
    b4[p] = acc;
    t1 += acc;
    t2 += acc * acc;
  }
  for (int m = 1; m < 32; m <<= 1) { t1 += __shfl_xor(t1, m, 32); t2 += __shfl_xor(t2, m, 32); }
  mu = t1 / 128.f;
  var = t2 / 128.f - mu * mu;
  rs = rsqrtf(var + EPS);
#pragma unroll
  for (int p = 0; p < 4; ++p) {
    int o = lane + p * 32;
    h2[o] = fmaxf((b4[p] - mu) * rs * hg2[t * 128 + o] + hbe2[t * 128 + o], 0.f);
  }
  __syncthreads();

  // layer3: 128 -> 3 ; u = delta - x_i
  float uval = 0.f;
  if (lane < 3) {
    const float* W3 = hw3 + t * 3 * 128;
    float acc = hb3[t * 3 + lane];
    for (int k = 0; k < 128; ++k) acc += W3[lane * 128 + k] * h2[k];
    uval = acc - xi[lane];
  }
  if (lane < 16) {
    uf[node * 16 + lane] = (f16)((lane < 3) ? uval : 0.f);
    float zc = (lane < 3) ? xi[lane] : ((lane < 11) ? si[lane - 3] : 0.f);
    zf[node * 16 + lane] = (f16)zc;
  }
}

// ---------------- edge kernel: per-wave node, WMMA edge MLP over adjacent j-tiles,
// masked max-pool, then g-MLP + residual --------------------------------------------------
__global__ __launch_bounds__(256) void kedge_PointGNN(
    const float* __restrict__ cur, const float* __restrict__ orig, const int* __restrict__ fsz,
    const f16* __restrict__ w1g, const f16* __restrict__ w2g, const f16* __restrict__ w3g,
    const float* __restrict__ fb1, const float* __restrict__ fg1, const float* __restrict__ fbe1,
    const float* __restrict__ fb2, const float* __restrict__ fg2, const float* __restrict__ fbe2,
    const float* __restrict__ fb3, const float* __restrict__ gw1, const float* __restrict__ gb1,
    const float* __restrict__ gg1, const float* __restrict__ gbe1, const float* __restrict__ gw2,
    const float* __restrict__ gb2, const float* __restrict__ gg2, const float* __restrict__ gbe2,
    const float* __restrict__ gw3, const float* __restrict__ gb3, const f16* __restrict__ zf,
    const f16* __restrict__ uf, float* __restrict__ nxt, int t) {
  extern __shared__ char smem[];
  f16* W1 = (f16*)smem;            // 64*32  f16
  f16* W2 = W1 + 64 * 32;          // 128*64 f16
  f16* W3 = W2 + 128 * 64;         // 128*128 f16
  float* P = (float*)(W3 + 128 * 128);
  float* pb1 = P;          float* pg1 = P + 64;  float* pbe1 = P + 128;
  float* pb2 = P + 192;    float* pg2 = P + 320; float* pbe2 = P + 448;
  float* pb3 = P + 576;    // 704 floats total
  char* wavebase = (char*)(P + 704);
  const int wv = threadIdx.x >> 5, lane = threadIdx.x & 31;
  char* my = wavebase + wv * 8192;
  f16* A0 = (f16*)my;              // 16x32 f16
  f16* H1 = A0 + 16 * 32;          // 16x64 f16
  f16* H2 = H1 + 16 * 64;          // 16x128 f16
  float* GS = (float*)(H2 + 16 * 128);  // 192 f32 (pool 128 + hidden 64)

  // cooperative weight staging (shared by all 8 waves)
  {
    const uint32_t* s1 = (const uint32_t*)(w1g + t * 64 * 32);
    uint32_t* d1 = (uint32_t*)W1;
    for (int k = threadIdx.x; k < 64 * 32 / 2; k += 256) d1[k] = s1[k];
    const uint32_t* s2 = (const uint32_t*)(w2g + t * 128 * 64);
    uint32_t* d2 = (uint32_t*)W2;
    for (int k = threadIdx.x; k < 128 * 64 / 2; k += 256) d2[k] = s2[k];
    const uint32_t* s3 = (const uint32_t*)(w3g + t * 128 * 128);
    uint32_t* d3 = (uint32_t*)W3;
    for (int k = threadIdx.x; k < 128 * 128 / 2; k += 256) d3[k] = s3[k];
    for (int k = threadIdx.x; k < 64; k += 256) {
      pb1[k] = fb1[t * 64 + k]; pg1[k] = fg1[t * 64 + k]; pbe1[k] = fbe1[t * 64 + k];
    }
    for (int k = threadIdx.x; k < 128; k += 256) {
      pb2[k] = fb2[t * 128 + k]; pg2[k] = fg2[t * 128 + k]; pbe2[k] = fbe2[t * 128 + k];
      pb3[k] = fb3[t * 128 + k];
    }
  }
  __syncthreads();

  const int node = blockIdx.x * 8 + wv;  // wave-uniform
  const int b = node / NN, i = node % NN;
  const int fs = fsz[b];
  const float* xi = orig + node * SD;
  const float xix = xi[0], xiy = xi[1], xiz = xi[2];
  const int col = lane & 15, hfl = lane >> 4;

  // zero the constant K=16..31 half of A0 once (layer-1 K pad)
  if (hfl == 1) {
    uint32_t* p = (uint32_t*)(A0 + (lane - 16) * 32 + 16);
#pragma unroll
    for (int q = 0; q < 8; ++q) p[q] = 0u;
  }
  v16h ureg = *(const v16h*)(uf + node * 16);
  float pool[8];
#pragma unroll
  for (int q = 0; q < 8; ++q) pool[q] = 0.f;

  if (i < fs) {  // wave-uniform: padded node -> no adjacency at all
    for (int j0 = 0; j0 < NN; j0 += 16) {
      const int j = j0 + col;
      unsigned mask;
      {
        const float* xj = orig + (b * NN + j) * SD;
        float dx = xix - xj[0], dy = xiy - xj[1], dz = xiz - xj[2];
        float d2 = dx * dx + dy * dy + dz * dz;
        bool adj = (hfl == 0) && (d2 < RR) && (j < fs);
        mask = (unsigned)(__ballot(adj) & 0xffffull);
      }
      if (!mask) continue;  // sparse skip (wave-uniform)

      // build A0 rows: eij_in = z_j + u_i (packed f16 add), cols 11..15 are z/u zeros
      if (hfl == 0) {
        v16h zr = *(const v16h*)(zf + (b * NN + j) * 16);
        *(v16h*)(A0 + col * 32) = zr + ureg;
      }
      LDS_FENCE();

      // layer 1: K=32 (11 live), N=64 -> 4 WMMA
      v8f acc1[4];
      {
        v16h Av = ldA(A0 + col * 32 + hfl * 8);
#pragma unroll
        for (int nt = 0; nt < 4; ++nt)
          acc1[nt] = wmma_f16(Av, ldB(W1 + (nt * 16 + col) * 32 + hfl * 8), vzero8());
      }
      ln_relu_store<4>(acc1, pb1, pg1, pbe1, H1, lane);
      LDS_FENCE();

      // layer 2: K=64, N=128 -> 16 WMMA
      v8f acc2[8];
#pragma unroll
      for (int nt = 0; nt < 8; ++nt) acc2[nt] = vzero8();
#pragma unroll
      for (int ks = 0; ks < 2; ++ks) {
        v16h Av = ldA(H1 + col * 64 + ks * 32 + hfl * 8);
#pragma unroll
        for (int nt = 0; nt < 8; ++nt)
          acc2[nt] = wmma_f16(Av, ldB(W2 + (nt * 16 + col) * 64 + ks * 32 + hfl * 8), acc2[nt]);
      }
      ln_relu_store<8>(acc2, pb2, pg2, pbe2, H2, lane);
      LDS_FENCE();

      // layer 3: K=128, N=128 -> 32 WMMA, relu + adjacency mask + max-pool (in registers)
      v8f acc3[8];
#pragma unroll
      for (int nt = 0; nt < 8; ++nt) acc3[nt] = vzero8();
#pragma unroll
      for (int ks = 0; ks < 4; ++ks) {
        v16h Av = ldA(H2 + col * 128 + ks * 32 + hfl * 8);
#pragma unroll
        for (int nt = 0; nt < 8; ++nt)
          acc3[nt] = wmma_f16(Av, ldB(W3 + (nt * 16 + col) * 128 + ks * 32 + hfl * 8), acc3[nt]);
      }
#pragma unroll
      for (int nt = 0; nt < 8; ++nt) {
        float bf = pb3[nt * 16 + col];
        float m = 0.f;
#pragma unroll
        for (int r = 0; r < 8; ++r) {
          int edge = hfl * 8 + r;  // D layout: VGPR r, low lanes edge r, high lanes edge r+8
          float v = ((mask >> edge) & 1u) ? fmaxf(acc3[nt][r] + bf, 0.f) : 0.f;
          m = fmaxf(m, v);
        }
        m = fmaxf(m, __shfl_xor(m, 16, 32));
        pool[nt] = fmaxf(pool[nt], m);
      }
    }
  }

  // ---- g MLP on pooled features + residual (wave-cooperative VALU) ----
  float* PO = GS;        // 128 floats
  float* HG = GS + 128;  // 64 floats
  if (hfl == 0) {
#pragma unroll
    for (int q = 0; q < 8; ++q) PO[q * 16 + col] = pool[q];
  }
  LDS_FENCE();
  const float* GW1 = gw1 + t * 64 * 128;
  float g1v[2];
  float s1 = 0.f, s2 = 0.f;
#pragma unroll
  for (int p = 0; p < 2; ++p) {
    int o = lane + p * 32;
    float acc = gb1[t * 64 + o];
    for (int k = 0; k < 128; ++k) acc += GW1[o * 128 + k] * PO[k];
    g1v[p] = acc;
    s1 += acc;
    s2 += acc * acc;
  }
  for (int m = 1; m < 32; m <<= 1) { s1 += __shfl_xor(s1, m, 32); s2 += __shfl_xor(s2, m, 32); }
  float mu = s1 / 64.f, var = s2 / 64.f - mu * mu, rs = rsqrtf(var + EPS);
#pragma unroll
  for (int p = 0; p < 2; ++p) {
    int o = lane + p * 32;
    HG[o] = fmaxf((g1v[p] - mu) * rs * gg1[t * 64 + o] + gbe1[t * 64 + o], 0.f);
  }
  LDS_FENCE();
  const float* GW2 = gw2 + t * 32 * 64;
  float acc2s = gb2[t * 32 + lane];
  for (int k = 0; k < 64; ++k) acc2s += GW2[lane * 64 + k] * HG[k];
  s1 = acc2s;
  s2 = acc2s * acc2s;
  for (int m = 1; m < 32; m <<= 1) { s1 += __shfl_xor(s1, m, 32); s2 += __shfl_xor(s2, m, 32); }
  mu = s1 / 32.f;
  var = s2 / 32.f - mu * mu;
  rs = rsqrtf(var + EPS);
  float hv = fmaxf((acc2s - mu) * rs * gg2[t * 32 + lane] + gbe2[t * 32 + lane], 0.f);
  PO[lane] = hv;  // reuse pool region (dead)
  LDS_FENCE();
  if (lane < 8) {
    const float* GW3 = gw3 + t * 8 * 32;
    float acc = gb3[t * 8 + lane];
    for (int k = 0; k < 32; ++k) acc += GW3[lane * 32 + k] * PO[k];
    nxt[node * SD + lane] = acc + cur[node * SD + lane];  // residual
  }
}

// ------------------------------------ host launcher --------------------------------------
extern "C" void kernel_launch(void* const* d_in, const int* in_sizes, int n_in, void* d_out,
                              int out_size, void* d_ws, size_t ws_size, hipStream_t stream) {
  (void)in_sizes; (void)n_in; (void)out_size; (void)ws_size;
  const float* st  = (const float*)d_in[0];
  const int*   fsz = (const int*)d_in[1];
  const float *h_w1 = (const float*)d_in[2],  *h_b1 = (const float*)d_in[3],
              *h_g1 = (const float*)d_in[4],  *h_be1 = (const float*)d_in[5];
  const float *h_w2 = (const float*)d_in[6],  *h_b2 = (const float*)d_in[7],
              *h_g2 = (const float*)d_in[8],  *h_be2 = (const float*)d_in[9];
  const float *h_w3 = (const float*)d_in[10], *h_b3 = (const float*)d_in[11];
  const float *f_w1 = (const float*)d_in[12], *f_b1 = (const float*)d_in[13],
              *f_g1 = (const float*)d_in[14], *f_be1 = (const float*)d_in[15];
  const float *f_w2 = (const float*)d_in[16], *f_b2 = (const float*)d_in[17],
              *f_g2 = (const float*)d_in[18], *f_be2 = (const float*)d_in[19];
  const float *f_w3 = (const float*)d_in[20], *f_b3 = (const float*)d_in[21];
  const float *g_w1 = (const float*)d_in[22], *g_b1 = (const float*)d_in[23],
              *g_g1 = (const float*)d_in[24], *g_be1 = (const float*)d_in[25];
  const float *g_w2 = (const float*)d_in[26], *g_b2 = (const float*)d_in[27],
              *g_g2 = (const float*)d_in[28], *g_be2 = (const float*)d_in[29];
  const float *g_w3 = (const float*)d_in[30], *g_b3 = (const float*)d_in[31];
  float* out = (float*)d_out;

  // workspace layout
  char* ws = (char*)d_ws;
  float* stA = (float*)ws;                   // B*N*SD f32
  float* stB = stA + BB * NN * SD;           // B*N*SD f32
  f16* zf  = (f16*)(stB + BB * NN * SD);     // B*N*16 f16
  f16* uff = zf + BB * NN * 16;              // B*N*16 f16
  f16* w1f = uff + BB * NN * 16;             // L*64*32
  f16* w2f = w1f + LL * 64 * 32;             // L*128*64
  f16* w3f = w2f + LL * 128 * 64;            // L*128*128

  const int cvt_total = LL * 64 * 32 + LL * 128 * 64 + LL * 128 * 128;
  kcvt_PointGNN<<<(cvt_total + 255) / 256, 256, 0, stream>>>(f_w1, f_w2, f_w3, w1f, w2f, w3f);

  constexpr int SMEM_EDGE = (64 * 32 + 128 * 64 + 128 * 128) * 2 + 704 * 4 + 8 * 8192;
  const float* curp = st;
  float* nxts[LL] = {stA, stB, out};
  for (int t = 0; t < LL; ++t) {
    knode_PointGNN<<<BB * NN, 32, 0, stream>>>(curp, st, h_w1, h_b1, h_g1, h_be1, h_w2, h_b2,
                                               h_g2, h_be2, h_w3, h_b3, zf, uff, t);
    kedge_PointGNN<<<BB * NN / 8, 256, SMEM_EDGE, stream>>>(
        curp, st, fsz, w1f, w2f, w3f, f_b1, f_g1, f_be1, f_b2, f_g2, f_be2, f_b3, g_w1, g_b1,
        g_g1, g_be1, g_w2, g_b2, g_g2, g_be2, g_w3, g_b3, zf, uff, nxts[t], t);
    curp = nxts[t];
  }
}
